// SchNet_266287973048
// MI455X (gfx1250) — matrix operand order repaired and checked by reading, not compiled, not run on hardware
//
#include <hip/hip_runtime.h>
#include <stdint.h>

// ---------------- WMMA types ----------------
typedef __attribute__((ext_vector_type(16))) __bf16 v16bf;
typedef __attribute__((ext_vector_type(8)))  float  v8f;

union Frag16 { v16bf v; uint4 q[2]; };

// ---------------- CDNA5 async global->LDS path (guarded) ----------------
#if defined(__AMDGCN__) && __has_builtin(__builtin_amdgcn_global_load_async_to_lds_b128)
#define HAVE_ASYNC_LDS 1
#else
#define HAVE_ASYNC_LDS 0
#endif

#if HAVE_ASYNC_LDS
// Builtin prototype (from the round-2 diagnostic) takes pointers to
// int-vector(16B): (v4i addrspace(1)*, v4i addrspace(3)*, imm offset, imm cpol)
typedef int v4i_ls __attribute__((vector_size(16)));
typedef __attribute__((address_space(1))) v4i_ls* gv4i_p;
typedef __attribute__((address_space(3))) v4i_ls* lv4i_p;

__device__ __forceinline__ void async_ld_b128(const void* g, void* l) {
  __builtin_amdgcn_global_load_async_to_lds_b128(
      (gv4i_p)g, (lv4i_p)l, 0 /*imm offset*/, 0 /*cpol*/);
}
__device__ __forceinline__ void wait_async0() {
#if __has_builtin(__builtin_amdgcn_s_wait_asynccnt)
  __builtin_amdgcn_s_wait_asynccnt(0);
#else
  asm volatile("s_wait_asynccnt 0x0" ::: "memory");
#endif
}
#endif

__device__ __forceinline__ unsigned short f2bf(float x) {
  union { float f; unsigned int u; } c; c.f = x;
  unsigned int r = c.u + 0x7FFFu + ((c.u >> 16) & 1u);   // round-to-nearest-even
  return (unsigned short)(r >> 16);
}

__device__ __forceinline__ float sspf(float x) {
  // softplus(x) - log(2), overflow-safe
  return fmaxf(x, 0.f) + __logf(1.f + __expf(-fabsf(x))) - 0.69314718f;
}

#define CUTOFF_F 10.0f
#define NG_K 100

// =====================================================================
// Fused edge pipeline (one interaction block):
//   gaussians -> GEMM1(W1)+b1 -> ssp -> GEMM2(W2)+b2 -> gate
//   -> * xf[src] -> atomicAdd agg[dst]
// Tile: 64 edges x 128 features per workgroup, 256 threads (8 wave32).
// Weight tiles streamed via async global->LDS DMA, overlapped with the
// gaussian build (GEMM1) and the ssp epilogue (GEMM2).
// =====================================================================
#define E_AST 136   // ushort stride, 272B rows (16B aligned, conflict-padded)
#define E_BST 136

__global__ __launch_bounds__(256)
void edge_block_kernel(const float* __restrict__ elen,
                       const int* __restrict__ srcI, const int* __restrict__ dstI,
                       const unsigned short* __restrict__ W1t,  // [128n][128k] bf16
                       const float* __restrict__ b1,
                       const unsigned short* __restrict__ W2t,  // [128n][128k] bf16
                       const float* __restrict__ b2,
                       const float* __restrict__ xf,            // [Nn][128] f32
                       float* __restrict__ agg,                 // [Nn][128] f32
                       int E) {
  __shared__ __align__(16) unsigned short As[64 * E_AST];
  __shared__ __align__(16) unsigned short Bs[128 * E_BST];
  __shared__ float dv[64];
  __shared__ int   sS[64], sD[64];

  const int tid  = threadIdx.x;
  const int lane = tid & 31;
  const int w    = tid >> 5;
  const int e0   = blockIdx.x * 64;

  if (tid < 64) {
    int e = e0 + tid;
    dv[tid] = (e < E) ? elen[e] : 1e9f;
    sS[tid] = (e < E) ? srcI[e] : 0;
    sD[tid] = (e < E) ? dstI[e] : 0;
  }
  __syncthreads();

#if HAVE_ASYNC_LDS
  // Kick off W1t DMA into Bs; overlaps with gaussian A-tile build below.
  for (int i = tid; i < (128 * 128) / 8; i += 256) {
    int idx8 = i * 8, r = idx8 >> 7, c = idx8 & 127;
    async_ld_b128(&W1t[idx8], &Bs[r * E_BST + c]);
  }
#endif

  // Build gaussian A-tile directly (fused smearing), zero-pad k in [100,128)
  const float dg    = CUTOFF_F / (float)(NG_K - 1);
  const float coeff = -0.5f / (dg * dg);
  for (int i = tid; i < 64 * 128; i += 256) {
    int r = i >> 7, k = i & 127;
    float v = 0.f;
    if (k < NG_K) { float t = dv[r] - (float)k * dg; v = __expf(coeff * t * t); }
    As[r * E_AST + k] = f2bf(v);
  }

#if HAVE_ASYNC_LDS
  wait_async0();
#else
  for (int i = tid; i < (128 * 128) / 8; i += 256) {
    int idx8 = i * 8, r = idx8 >> 7, c = idx8 & 127;
    *(uint4*)&Bs[r * E_BST + c] = *(const uint4*)&W1t[idx8];
  }
#endif
  __syncthreads();

  const int rs    = (w & 3) * 16;   // row strip in tile
  const int nb    = (w >> 2) * 64;  // col base
  const int khalf = lane >> 4;
  const int mrow  = rs + (lane & 15);

  v8f acc[4];
  #pragma unroll
  for (int t = 0; t < 4; t++) acc[t] = (v8f){0,0,0,0,0,0,0,0};

  // ---- GEMM1: [64x128] @ W1t -> [64x128] ----
  #pragma unroll
  for (int kk = 0; kk < 128; kk += 32) {
    Frag16 a;
    a.q[0] = *(const uint4*)&As[mrow * E_AST + kk + khalf * 8];
    a.q[1] = *(const uint4*)&As[mrow * E_AST + kk + 16 + khalf * 8];
    #pragma unroll
    for (int t = 0; t < 4; t++) {
      int n = nb + t * 16 + (lane & 15);
      Frag16 b;
      b.q[0] = *(const uint4*)&Bs[n * E_BST + kk + khalf * 8];
      b.q[1] = *(const uint4*)&Bs[n * E_BST + kk + 16 + khalf * 8];
      acc[t] = __builtin_amdgcn_wmma_f32_16x16x32_bf16(
          false, a.v, false, b.v, (short)0, acc[t], false, false);
    }
  }
  __syncthreads();

#if HAVE_ASYNC_LDS
  // Kick off W2t DMA into Bs; overlaps with the ssp epilogue below.
  for (int i = tid; i < (128 * 128) / 8; i += 256) {
    int idx8 = i * 8, r = idx8 >> 7, c = idx8 & 127;
    async_ld_b128(&W2t[idx8], &Bs[r * E_BST + c]);
  }
#endif

  // bias + ssp -> mid tile back into As (bf16)
  #pragma unroll
  for (int t = 0; t < 4; t++) {
    int col = nb + t * 16 + (lane & 15);
    float bv = b1[col];
    #pragma unroll
    for (int j = 0; j < 8; j++) {
      int row = rs + j + 8 * khalf;
      As[row * E_AST + col] = f2bf(sspf(acc[t][j] + bv));
    }
  }

#if HAVE_ASYNC_LDS
  wait_async0();
#else
  for (int i = tid; i < (128 * 128) / 8; i += 256) {
    int idx8 = i * 8, r = idx8 >> 7, c = idx8 & 127;
    *(uint4*)&Bs[r * E_BST + c] = *(const uint4*)&W2t[idx8];
  }
#endif
  __syncthreads();

  v8f acc2[4];
  #pragma unroll
  for (int t = 0; t < 4; t++) acc2[t] = (v8f){0,0,0,0,0,0,0,0};

  // ---- GEMM2: mid @ W2t -> filter W [64x128] ----
  #pragma unroll
  for (int kk = 0; kk < 128; kk += 32) {
    Frag16 a;
    a.q[0] = *(const uint4*)&As[mrow * E_AST + kk + khalf * 8];
    a.q[1] = *(const uint4*)&As[mrow * E_AST + kk + 16 + khalf * 8];
    #pragma unroll
    for (int t = 0; t < 4; t++) {
      int n = nb + t * 16 + (lane & 15);
      Frag16 b;
      b.q[0] = *(const uint4*)&Bs[n * E_BST + kk + khalf * 8];
      b.q[1] = *(const uint4*)&Bs[n * E_BST + kk + 16 + khalf * 8];
      acc2[t] = __builtin_amdgcn_wmma_f32_16x16x32_bf16(
          false, a.v, false, b.v, (short)0, acc2[t], false, false);
    }
  }

  // ---- epilogue: bias, gate, message = W * xf[src], scatter-add to agg[dst]
  #pragma unroll
  for (int t = 0; t < 4; t++) {
    int col = nb + t * 16 + (lane & 15);
    float bv = b2[col];
    #pragma unroll
    for (int j = 0; j < 8; j++) {
      int r = rs + j + 8 * khalf;
      int e = e0 + r;
      if (e < E) {
        float gate = (dv[r] <= CUTOFF_F) ? 1.f : 0.f;
        float wv = (acc2[t][j] + bv) * gate;
        float m  = wv * xf[(long)sS[r] * 128 + col];
        atomicAdd(&agg[(long)sD[r] * 128 + col], m);
      }
    }
  }
}

// =====================================================================
// Generic node-side WMMA GEMM:
//   out = act( A_bf16[MxK] @ Bt_bf16[N][K] + bias[N] )
//   act: 0=none 1=ssp 2=relu ; accum: outF += result (residual)
//   outF (f32) and/or outB (bf16) written.
// Tile 64x128, 256 threads, K staged in 32-wide chunks.
// Async path: double-buffered LDS with ASYNCcnt-pipelined prefetch.
// =====================================================================
#define N_AST 48

__global__ __launch_bounds__(256)
void gemm_node_kernel(const unsigned short* __restrict__ A,
                      const unsigned short* __restrict__ Bt,
                      const float* __restrict__ bias,
                      float* __restrict__ outF,
                      unsigned short* __restrict__ outB,
                      int M, int N, int K, int act, int accum) {
  __shared__ __align__(16) unsigned short As[2][64 * N_AST];
  __shared__ __align__(16) unsigned short Bs[2][128 * N_AST];

  const int tid  = threadIdx.x;
  const int lane = tid & 31;
  const int w    = tid >> 5;
  const int m0   = blockIdx.x * 64;
  const int n0   = blockIdx.y * 128;
  const int rs    = (w & 3) * 16;
  const int nb    = (w >> 2) * 64;
  const int khalf = lane >> 4;
  const int mrow  = rs + (lane & 15);

  v8f acc[4];
  #pragma unroll
  for (int t = 0; t < 4; t++) acc[t] = (v8f){0,0,0,0,0,0,0,0};

  auto computeChunk = [&](int buf) {
    const unsigned short* as = As[buf];
    const unsigned short* bs = Bs[buf];
    Frag16 a;
    a.q[0] = *(const uint4*)&as[mrow * N_AST + khalf * 8];
    a.q[1] = *(const uint4*)&as[mrow * N_AST + 16 + khalf * 8];
    #pragma unroll
    for (int t = 0; t < 4; t++) {
      int n = nb + t * 16 + (lane & 15);
      Frag16 b;
      b.q[0] = *(const uint4*)&bs[n * N_AST + khalf * 8];
      b.q[1] = *(const uint4*)&bs[n * N_AST + 16 + khalf * 8];
      acc[t] = __builtin_amdgcn_wmma_f32_16x16x32_bf16(
          false, a.v, false, b.v, (short)0, acc[t], false, false);
    }
  };

#if HAVE_ASYNC_LDS
  auto stageAsync = [&](int buf, int kk) {
    {
      int idx8 = tid * 8;
      int r = idx8 >> 5, c = idx8 & 31;
      int gm = m0 + r;
      if (gm >= M) gm = M - 1;  // clamp: feeds only discarded C rows
      async_ld_b128(&A[(long)gm * K + kk + c], &As[buf][r * N_AST + c]);
    }
    #pragma unroll
    for (int p = 0; p < 2; p++) {
      int idx8 = (tid + p * 256) * 8;
      int r = idx8 >> 5, c = idx8 & 31;
      async_ld_b128(&Bt[(long)(n0 + r) * K + kk + c], &Bs[buf][r * N_AST + c]);
    }
  };

  stageAsync(0, 0);
  wait_async0();
  __syncthreads();
  int cur = 0;
  for (int kk = 0; kk < K; kk += 32) {
    if (kk + 32 < K) stageAsync(cur ^ 1, kk + 32);  // prefetch next chunk
    computeChunk(cur);
    wait_async0();
    __syncthreads();
    cur ^= 1;
  }
#else
  for (int kk = 0; kk < K; kk += 32) {
    {
      int idx8 = tid * 8;
      int r = idx8 >> 5, c = idx8 & 31;
      int gm = m0 + r;
      uint4 val = {0u, 0u, 0u, 0u};
      if (gm < M) val = *(const uint4*)&A[(long)gm * K + kk + c];
      *(uint4*)&As[0][r * N_AST + c] = val;
    }
    #pragma unroll
    for (int p = 0; p < 2; p++) {
      int idx8 = (tid + p * 256) * 8;
      int r = idx8 >> 5, c = idx8 & 31;
      *(uint4*)&Bs[0][r * N_AST + c] = *(const uint4*)&Bt[(long)(n0 + r) * K + kk + c];
    }
    __syncthreads();
    computeChunk(0);
    __syncthreads();
  }
#endif

  #pragma unroll
  for (int t = 0; t < 4; t++) {
    int gn = n0 + nb + t * 16 + (lane & 15);
    float bv = bias ? bias[gn] : 0.f;
    #pragma unroll
    for (int j = 0; j < 8; j++) {
      int gm = m0 + rs + j + 8 * khalf;
      if (gm < M) {
        float x = acc[t][j] + bv;
        if (act == 1) x = sspf(x);
        else if (act == 2) x = fmaxf(x, 0.f);
        long idx = (long)gm * N + gn;
        if (accum) { x += outF[idx]; outF[idx] = x; }
        else if (outF) outF[idx] = x;
        if (outB) outB[idx] = f2bf(x);
      }
    }
  }
}

// =====================================================================
// Small helper kernels
// =====================================================================
__global__ void pack_weight_kernel(const float* __restrict__ W,  // [K][N] row-major
                                   unsigned short* __restrict__ Wt, // [N][Kpad]
                                   int K, int N, int Kpad) {
  int idx = blockIdx.x * blockDim.x + threadIdx.x;
  if (idx >= N * Kpad) return;
  int n = idx / Kpad, k = idx % Kpad;
  float v = (k < K) ? W[(long)k * N + n] : 0.f;
  Wt[idx] = f2bf(v);
}

__global__ void embed_kernel(const int* __restrict__ types,
                             const float* __restrict__ emb,
                             float* __restrict__ h, unsigned short* __restrict__ hb,
                             int total) {
  int idx = blockIdx.x * blockDim.x + threadIdx.x;
  if (idx >= total) return;
  int n = idx >> 8, c = idx & 255;
  float v = emb[(long)types[n] * 256 + c];
  h[idx] = v;
  hb[idx] = f2bf(v);
}

__global__ void fill_f32_kernel(float* __restrict__ p, float v, int n) {
  int idx = blockIdx.x * blockDim.x + threadIdx.x;
  if (idx < n) p[idx] = v;
}

__global__ void f32_to_bf16_kernel(const float* __restrict__ s,
                                   unsigned short* __restrict__ d, int n) {
  int idx = blockIdx.x * blockDim.x + threadIdx.x;
  if (idx < n) d[idx] = f2bf(s[idx]);
}

__device__ __forceinline__ void atomicMaxFloat(float* addr, float val) {
  unsigned int* ua = (unsigned int*)addr;
  unsigned int old = *ua;
  while (__uint_as_float(old) < val) {
    unsigned int assumed = old;
    old = atomicCAS(ua, assumed, __float_as_uint(val));
    if (old == assumed) break;
  }
}

__global__ void pool_max_kernel(const float* __restrict__ h,
                                const int* __restrict__ seg,
                                float* __restrict__ pooled, int total) {
  int idx = blockIdx.x * blockDim.x + threadIdx.x;
  if (idx >= total) return;
  int n = idx >> 8, c = idx & 255;
  atomicMaxFloat(&pooled[(long)seg[n] * 256 + c], h[idx]);
}

// =====================================================================
// Host launcher
// =====================================================================
extern "C" void kernel_launch(void* const* d_in, const int* in_sizes, int n_in,
                              void* d_out, int out_size, void* d_ws, size_t ws_size,
                              hipStream_t stream) {
  (void)n_in; (void)out_size; (void)ws_size;
  const int*   atom_types = (const int*)  d_in[0];
  const int*   edge_index = (const int*)  d_in[1];
  const float* edge_len   = (const float*)d_in[2];
  const int*   batch_seg  = (const int*)  d_in[3];
  const float* atom_emb   = (const float*)d_in[4];
  const float* W1s    = (const float*)d_in[5];
  const float* b1s    = (const float*)d_in[6];
  const float* W2s    = (const float*)d_in[7];
  const float* b2s    = (const float*)d_in[8];
  const float* lin1Ws = (const float*)d_in[9];
  const float* lin2Ws = (const float*)d_in[10];
  const float* lin2bs = (const float*)d_in[11];
  const float* linWs  = (const float*)d_in[12];
  const float* linbs  = (const float*)d_in[13];
  const float* fW1 = (const float*)d_in[14];
  const float* fb1 = (const float*)d_in[15];
  const float* fW2 = (const float*)d_in[16];
  const float* fb2 = (const float*)d_in[17];

  const int Nn = in_sizes[0];      // 20000 nodes
  const int E  = in_sizes[2];      // 640000 edges
  const int*  srcI = edge_index;
  const int*  dstI = edge_index + E;

  // ---- workspace carve-out (256B aligned regions) ----
  char* p = (char*)d_ws;
  auto carve = [&](size_t bytes) -> void* {
    void* r = (void*)p;
    p += (bytes + 255) & ~(size_t)255;
    return r;
  };
  float*          h       = (float*)         carve((size_t)Nn * 256 * 4);
  unsigned short* hb      = (unsigned short*)carve((size_t)Nn * 256 * 2);
  float*          xf      = (float*)         carve((size_t)Nn * 128 * 4);
  float*          agg     = (float*)         carve((size_t)Nn * 128 * 4);
  unsigned short* aggb    = (unsigned short*)carve((size_t)Nn * 128 * 2);
  unsigned short* hcb     = (unsigned short*)carve((size_t)Nn * 256 * 2);
  float*          pooled  = (float*)         carve((size_t)64 * 256 * 4);
  unsigned short* pooledb = (unsigned short*)carve((size_t)64 * 256 * 2);
  unsigned short* tb      = (unsigned short*)carve((size_t)64 * 256 * 2);
  unsigned short* W1t = (unsigned short*)carve((size_t)6 * 128 * 128 * 2);
  unsigned short* W2t = (unsigned short*)carve((size_t)6 * 128 * 128 * 2);
  unsigned short* l1t = (unsigned short*)carve((size_t)6 * 128 * 256 * 2);
  unsigned short* l2t = (unsigned short*)carve((size_t)6 * 256 * 128 * 2);
  unsigned short* lt  = (unsigned short*)carve((size_t)6 * 256 * 256 * 2);
  unsigned short* f1t = (unsigned short*)carve((size_t)256 * 256 * 2);
  unsigned short* f2t = (unsigned short*)carve((size_t)256 * 256 * 2);

  auto pack = [&](const float* W, unsigned short* Wt, int K, int N, int Kpad) {
    int cnt = N * Kpad;
    pack_weight_kernel<<<(cnt + 255) / 256, 256, 0, stream>>>(W, Wt, K, N, Kpad);
  };

  // ---- pack all weights to bf16, transposed [N][Kpad] ----
  for (int i = 0; i < 6; i++) {
    pack(W1s    + (long)i * 100 * 128, W1t + (long)i * 128 * 128, 100, 128, 128);
    pack(W2s    + (long)i * 128 * 128, W2t + (long)i * 128 * 128, 128, 128, 128);
    pack(lin1Ws + (long)i * 256 * 128, l1t + (long)i * 128 * 256, 256, 128, 256);
    pack(lin2Ws + (long)i * 128 * 256, l2t + (long)i * 256 * 128, 128, 256, 128);
    pack(linWs  + (long)i * 256 * 256, lt  + (long)i * 256 * 256, 256, 256, 256);
  }
  pack(fW1, f1t, 256, 256, 256);
  pack(fW2, f2t, 256, 256, 256);

  // ---- embedding ----
  {
    int cnt = Nn * 256;
    embed_kernel<<<(cnt + 255) / 256, 256, 0, stream>>>(atom_types, atom_emb, h, hb, cnt);
  }

  const int mg = (Nn + 63) / 64;         // node-GEMM row tiles
  const int eg = (E + 63) / 64;          // edge tiles

  // ---- interaction blocks ----
  for (int i = 0; i < 6; i++) {
    // xf = h @ lin1W  [Nn x 128]
    gemm_node_kernel<<<dim3(mg, 1), 256, 0, stream>>>(
        hb, l1t + (long)i * 128 * 256, nullptr, xf, nullptr,
        Nn, 128, 256, 0, 0);
    // agg = 0
    {
      int cnt = Nn * 128;
      fill_f32_kernel<<<(cnt + 255) / 256, 256, 0, stream>>>(agg, 0.f, cnt);
    }
    // fused edge pipeline: gaussian -> MLP -> gate -> msg -> scatter
    edge_block_kernel<<<eg, 256, 0, stream>>>(
        edge_len, srcI, dstI,
        W1t + (long)i * 128 * 128, b1s + (long)i * 128,
        W2t + (long)i * 128 * 128, b2s + (long)i * 128,
        xf, agg, E);
    // agg -> bf16
    {
      int cnt = Nn * 128;
      f32_to_bf16_kernel<<<(cnt + 255) / 256, 256, 0, stream>>>(agg, aggb, cnt);
    }
    // hc = ssp(agg @ lin2W + lin2b)   [Nn x 256] (bf16)
    gemm_node_kernel<<<dim3(mg, 2), 256, 0, stream>>>(
        aggb, l2t + (long)i * 256 * 128, lin2bs + (long)i * 256, nullptr, hcb,
        Nn, 256, 128, 1, 0);
    // h += hc @ linW + linb ; refresh hb
    gemm_node_kernel<<<dim3(mg, 2), 256, 0, stream>>>(
        hcb, lt + (long)i * 256 * 256, linbs + (long)i * 256, h, hb,
        Nn, 256, 256, 0, 1);
  }

  // ---- global max pool ----
  {
    int cnt = 64 * 256;
    fill_f32_kernel<<<(cnt + 255) / 256, 256, 0, stream>>>(pooled, -3.0e38f, cnt);
  }
  {
    int cnt = Nn * 256;
    pool_max_kernel<<<(cnt + 255) / 256, 256, 0, stream>>>(h, batch_seg, pooled, cnt);
  }
  {
    int cnt = 64 * 256;
    f32_to_bf16_kernel<<<(cnt + 255) / 256, 256, 0, stream>>>(pooled, pooledb, cnt);
  }

  // ---- final MLP: relu(pooled@fW1+fb1)@fW2+fb2 -> d_out [64x256] f32 ----
  gemm_node_kernel<<<dim3(1, 2), 256, 0, stream>>>(
      pooledb, f1t, fb1, nullptr, tb, 64, 256, 256, 2, 0);
  gemm_node_kernel<<<dim3(1, 2), 256, 0, stream>>>(
      tb, f2t, fb2, (float*)d_out, nullptr, 64, 256, 256, 0, 0);
}
